// CRFWithConstraints_15006615734073
// MI455X (gfx1250) — matrix operand order repaired and checked by reading, not compiled, not run on hardware
//
#include <hip/hip_runtime.h>
#include <hip/hip_bf16.h>

typedef _Float16 v16h __attribute__((ext_vector_type(16)));
typedef _Float16 v8h  __attribute__((ext_vector_type(8)));
typedef float    v8f  __attribute__((ext_vector_type(8)));

#define TT   48      // number of tags
#define TPAD 64      // K padded to 2 WMMA chunks of 32
#define NEG_INF (-__builtin_inff())
#define LOG2E 1.4426950408889634f
#define LN2   0.6931471805599453f

// Raw hardware transcendentals on the device pass (args provably in safe
// range: no denorm/inf fixups needed -> single v_exp_f32 / v_log_f32).
// Host pass: plain libm names, which HIP declares as __device__ overloads,
// so the host semantic check of the kernel bodies passes.
#if defined(__HIP_DEVICE_COMPILE__) && __has_builtin(__builtin_amdgcn_exp2f)
#define EXP2F(x) __builtin_amdgcn_exp2f(x)
#else
#define EXP2F(x) exp2f(x)
#endif
#if defined(__HIP_DEVICE_COMPILE__) && __has_builtin(__builtin_amdgcn_logf)
#define LOG2F(x) __builtin_amdgcn_logf(x)
#else
#define LOG2F(x) log2f(x)
#endif

union V16U { v16h v; v8h h[2]; };

// ---------------------------------------------------------------------------
// Denominator: CRF forward algorithm. One wave = 16 batch rows.
// log-semiring matvec as f16 WMMA on exp-shifted scores:
//   next[b,j] = m_b + log( sum_i exp(sc[b,i]-m_b) * E[i,j] ) + em[s,b,j]
// E = exp(ct) is constant -> B fragments live in registers for the whole scan.
// ---------------------------------------------------------------------------
__global__ __launch_bounds__(32)
void crf_forward_scan_wmma(const float* __restrict__ emis,
                           const unsigned char* __restrict__ mask,
                           const float* __restrict__ trans,
                           const float* __restrict__ startt,
                           const float* __restrict__ endt,
                           const float* __restrict__ constr,
                           float* __restrict__ denom,
                           int B, int S)
{
    __shared__ __align__(16) float    sc[16][TT];    // current log-scores
    __shared__ __align__(16) _Float16 pA[16][TPAD];  // exp-shifted, A staging
    __shared__ __align__(16) _Float16 Eb[TPAD][TT];  // exp(ct), rows 48..63 = 0
    __shared__ __align__(16) float    mrow[16];      // per-row max
    __shared__ __align__(16) int      mflag[16];     // per-row mask this step
    __shared__ __align__(16) float    pmax[16][2];   // partial maxes (2 halves)

    const int tid = threadIdx.x;         // 0..31 (wave32)
    const int lh  = tid >> 4;            // lane half
    const int ln  = tid & 15;
    const int b0  = blockIdx.x * 16;     // first batch row of this wave

    // ---- prologue: E[i][j] = (constr==0) ? 0 : exp(trans*constr); pad rows 0
    for (int idx = tid; idx < TPAD * TT; idx += 32) {
        int k = idx / TT, j = idx % TT;
        float v = 0.0f;
        if (k < TT) {
            float c = constr[k * TT + j];
            v = (c == 0.0f) ? 0.0f : EXP2F(trans[k * TT + j] * c * LOG2E);
        }
        Eb[k][j] = (_Float16)v;
    }
    // zero the padded K columns of pA once (never rewritten)
    for (int idx = tid; idx < 16 * (TPAD - TT); idx += 32) {
        int r = idx / (TPAD - TT), c = TT + idx % (TPAD - TT);
        pA[r][c] = (_Float16)0.0f;
    }
    // init scores: sc[b][t] = start[t] + em[s=0,b,t]
    if (tid < 16) {
        const float* e0 = emis + ((size_t)(b0 + tid) * S) * TT;
        for (int t = 0; t < TT; ++t) sc[tid][t] = startt[t] + e0[t];
    }
    __syncthreads();

    // ---- loop-invariant B fragments: 3 N-tiles x 2 K-chunks, in VGPRs.
    // 16-bit B 32x16: VGPR r holds K=2r,2r+1 (lanes 0-15) / K=8+2r,+1
    // (lanes 16-31), both +8 if r>=4; N = lane%16.  v16h element e <-> K = c*32 + lh*8 + e (+8 if e>=8).
    v16h Bf[3][2];
    for (int n = 0; n < 3; ++n)
        for (int c = 0; c < 2; ++c) {
            const int col = n * 16 + ln;
            const int kb  = c * 32 + lh * 8;
            v16h b;
            #pragma unroll
            for (int e = 0; e < 16; ++e) b[e] = Eb[kb + e + (e >= 8 ? 8 : 0)][col];
            Bf[n][c] = b;
        }

    // per-lane emission base offsets for phase C rows (strength reduction)
    size_t rowoff[8];
    #pragma unroll
    for (int v = 0; v < 8; ++v)
        rowoff[v] = ((size_t)(b0 + v + lh * 8) * S) * TT + ln;

    size_t sTT = (size_t)TT;             // s * TT, maintained incrementally

    // ---- sequential scan over the sequence
    for (int s = 1; s < S; ++s) {
        // phase A1: each lane owns half a row (24 cols): partial max
        const int   r  = ln;             // row this lane helps with
        const float* srow = &sc[r][lh * 24];
        {
            const float4* s4 = (const float4*)srow;   // 24 floats = 6x float4
            float4 a = s4[0], b = s4[1], c = s4[2];
            float4 d = s4[3], e = s4[4], f = s4[5];
            float pm = fmaxf(fmaxf(fmaxf(a.x, a.y), fmaxf(a.z, a.w)),
                             fmaxf(fmaxf(b.x, b.y), fmaxf(b.z, b.w)));
            pm = fmaxf(pm, fmaxf(fmaxf(c.x, c.y), fmaxf(c.z, c.w)));
            pm = fmaxf(pm, fmaxf(fmaxf(d.x, d.y), fmaxf(d.z, d.w)));
            pm = fmaxf(pm, fmaxf(fmaxf(e.x, e.y), fmaxf(e.z, e.w)));
            pm = fmaxf(pm, fmaxf(fmaxf(f.x, f.y), fmaxf(f.z, f.w)));
            pmax[r][lh] = pm;
            if (lh == 0) {
                mflag[r] = mask[(size_t)(b0 + r) * S + s];
            } else if (s + 1 < S) {      // warm next step's emission row
                __builtin_prefetch(&emis[((size_t)(b0 + r) * S + s + 1) * TT], 0, 1);
            }
        }
        __syncthreads();

        // phase A2: full max; p = exp2((sc-mx)*log2e) packed to v8h stores
        {
            float mx = fmaxf(pmax[r][0], pmax[r][1]);
            if (lh == 0) mrow[r] = mx;
            _Float16* prow = &pA[r][lh * 24];
            #pragma unroll
            for (int c8 = 0; c8 < 3; ++c8) {
                v8h o;
                #pragma unroll
                for (int e = 0; e < 8; ++e)
                    o[e] = (_Float16)EXP2F((srow[c8 * 8 + e] - mx) * LOG2E);
                *(v8h*)&prow[c8 * 8] = o;
            }
        }
        __syncthreads();

        // phase B: A fragments (two ds_load_b128 per chunk), 6 chained WMMAs
        V16U A0, A1;
        {
            const int M = ln, o = lh * 8;
            A0.h[0] = *(const v8h*)&pA[M][o];
            A0.h[1] = *(const v8h*)&pA[M][16 + o];
            A1.h[0] = *(const v8h*)&pA[M][32 + o];
            A1.h[1] = *(const v8h*)&pA[M][48 + o];
        }
        const v8f zero = {};
        v8f acc[3];
        #pragma unroll
        for (int n = 0; n < 3; ++n) {
            v8f t0 = __builtin_amdgcn_wmma_f32_16x16x32_f16(
                false, A0.v, false, Bf[n][0], (short)0, zero, false, false);
            acc[n] = __builtin_amdgcn_wmma_f32_16x16x32_f16(
                false, A1.v, false, Bf[n][1], (short)0, t0, false, false);
        }

        // phase C prep: flags + row maxes for this lane's 8 rows, vector loads
        union { int4   q[2]; int   i[8]; } FL;
        union { float4 q[2]; float f[8]; } MR;
        FL.q[0] = *(const int4*)  &mflag[lh * 8];
        FL.q[1] = *(const int4*)  &mflag[lh * 8 + 4];
        MR.q[0] = *(const float4*)&mrow[lh * 8];
        MR.q[1] = *(const float4*)&mrow[lh * 8 + 4];

        // phase C: row = v + lh*8, col = n*16 + ln; predicate only the stores
        #pragma unroll
        for (int v = 0; v < 8; ++v) {
            const int    row = v + lh * 8;
            const float* ep  = emis + rowoff[v] + sTT;
            float e0 = ep[0], e1 = ep[16], e2 = ep[32];
            float n0 = MR.f[v] + LN2 * LOG2F(acc[0][v]) + e0;
            float n1 = MR.f[v] + LN2 * LOG2F(acc[1][v]) + e1;
            float n2 = MR.f[v] + LN2 * LOG2F(acc[2][v]) + e2;
            if (FL.i[v]) {
                sc[row][ln]      = n0;
                sc[row][16 + ln] = n1;
                sc[row][32 + ln] = n2;
            }
        }
        sTT += TT;
        __syncthreads();
    }

    // ---- finalize: denom[b] = logsumexp_t(sc[b][t] + end[t])
    if (tid < 16) {
        const int row = tid;
        float mx = NEG_INF;
        for (int t = 0; t < TT; ++t) mx = fmaxf(mx, sc[row][t] + endt[t]);
        float sum = 0.0f;
        for (int t = 0; t < TT; ++t)
            sum += EXP2F((sc[row][t] + endt[t] - mx) * LOG2E);
        denom[b0 + row] = mx + LN2 * LOG2F(sum);
    }
}

// ---------------------------------------------------------------------------
// Numerator: gold-path score, one thread per batch row (pure gathers).
// ---------------------------------------------------------------------------
__global__ void crf_numerator(const float* __restrict__ emis,
                              const int* __restrict__ tags,
                              const unsigned char* __restrict__ mask,
                              const float* __restrict__ trans,
                              const float* __restrict__ startt,
                              const float* __restrict__ endt,
                              const float* __restrict__ constr,
                              float* __restrict__ num, int B, int S)
{
    int b = blockIdx.x * blockDim.x + threadIdx.x;
    if (b >= B) return;
    const int*           tg = tags + (size_t)b * S;
    const unsigned char* mk = mask + (size_t)b * S;
    const float*         em = emis + (size_t)b * S * TT;

    int   prev = tg[0];
    float acc  = startt[prev] + em[prev];
    int   cnt  = mk[0] ? 1 : 0;
    for (int s = 1; s < S; ++s) {
        int t = tg[s];
        if (mk[s]) {
            float c  = constr[prev * TT + t];
            float ct = (c == 0.0f) ? NEG_INF : trans[prev * TT + t] * c;
            acc += ct + em[(size_t)s * TT + t];
            ++cnt;
        }
        prev = t;
    }
    acc += endt[tg[cnt - 1]];   // seq_ends = sum(mask) - 1
    num[b] = acc;
}

// ---------------------------------------------------------------------------
// Final: out = mean(denom - num) = -mean(llh)
// ---------------------------------------------------------------------------
__global__ __launch_bounds__(256)
void crf_reduce(const float* __restrict__ num, const float* __restrict__ den,
                float* __restrict__ out, int B)
{
    __shared__ float part[256];
    int tid = threadIdx.x;
    float s = 0.0f;
    for (int i = tid; i < B; i += 256) s += den[i] - num[i];
    part[tid] = s;
    __syncthreads();
    for (int off = 128; off > 0; off >>= 1) {
        if (tid < off) part[tid] += part[tid + off];
        __syncthreads();
    }
    if (tid == 0) out[0] = part[0] / (float)B;
}

extern "C" void kernel_launch(void* const* d_in, const int* in_sizes, int n_in,
                              void* d_out, int out_size, void* d_ws, size_t ws_size,
                              hipStream_t stream)
{
    const float*         emis   = (const float*)d_in[0];          // (B,S,T) f32
    const int*           tags   = (const int*)d_in[1];            // (B,S) i32
    const unsigned char* mask   = (const unsigned char*)d_in[2];  // (B,S) bool
    const float*         trans  = (const float*)d_in[3];          // (T,T)
    const float*         startt = (const float*)d_in[4];          // (T,)
    const float*         endt   = (const float*)d_in[5];          // (T,)
    const float*         constr = (const float*)d_in[6];          // (T,T)

    const int B = 1024, S = 1024;   // reference shapes (T=48 baked into kernels)

    float* num = (float*)d_ws;      // [B] numerator scores
    float* den = num + B;           // [B] denominator log-partition

    crf_numerator<<<(B + 255) / 256, 256, 0, stream>>>(
        emis, tags, mask, trans, startt, endt, constr, num, B, S);
    crf_forward_scan_wmma<<<B / 16, 32, 0, stream>>>(
        emis, mask, trans, startt, endt, constr, den, B, S);
    crf_reduce<<<1, 256, 0, stream>>>(num, den, (float*)d_out, B);
}
